// GRAPH_VAE_19610820673592
// MI455X (gfx1250) — matrix operand order repaired and checked by reading, not compiled, not run on hardware
//
#include <hip/hip_runtime.h>
#include <math.h>

typedef __attribute__((ext_vector_type(16))) _Float16 v16h;
typedef __attribute__((ext_vector_type(8)))  _Float16 v8h;
typedef __attribute__((ext_vector_type(8)))  float    v8f;

// ---------------- WMMA fragment helpers (gfx1250 wave32 layouts) ----------------
// A (16x32 f16), row-major with stride ld:
// lane L<16 holds row M=L, K=0..15; lane L+16 holds K=16..31.
// Each lane's data = two contiguous 8-half runs -> two 16B LDS loads.
static __device__ __forceinline__ v16h frag_a_lds(const _Float16* A, int ld) {
  const int lane = threadIdx.x & 31;
  const int m    = lane & 15;
  const int hb   = (lane >> 4) * 16;
  const _Float16* row = A + m * ld + hb;
  v8h lo = *(const v8h*)(row);        // K = hb .. hb+7
  v8h hi = *(const v8h*)(row + 8);    // K = hb+8 .. hb+15
  return __builtin_shufflevector(lo, hi, 0, 1, 2, 3, 4, 5, 6, 7,
                                        8, 9, 10, 11, 12, 13, 14, 15);
}

// B (32x16 f16) from an LDS tile stored TRANSPOSED as [N][K] with row stride ldk:
// lane L<16 -> N=L, K=0..15 ; lane L+16 -> N=L-16, K=16..31.
// Each lane reads 16 contiguous halves -> one 32B LDS load (2x ds_load_b128).
static __device__ __forceinline__ v16h frag_b_t(const _Float16* Bt, int ldk) {
  const int lane = threadIdx.x & 31;
  const int n    = lane & 15;
  const int kb   = (lane >> 4) * 16;
  return *(const v16h*)(Bt + n * ldk + kb);
}

// ---------------- small utility kernels ----------------
__global__ void k_zero(float* __restrict__ p, int n) {
  int i = blockIdx.x * 256 + threadIdx.x;
  if (i < n) p[i] = 0.0f;
}

// agg[dst[e]] += h[src[e]]  (one thread per edge-feature)
__global__ void k_agg(const float* __restrict__ h, const int* __restrict__ src,
                      const int* __restrict__ dst, float* __restrict__ agg,
                      int E, int D) {
  int id = blockIdx.x * 256 + threadIdx.x;
  if (id >= E * D) return;
  int e = id / D, f = id - e * D;
  atomicAdd(&agg[(size_t)dst[e] * D + f], h[(size_t)src[e] * D + f]);
}

__global__ void k_pool(const float* __restrict__ h, const int* __restrict__ batch,
                       float* __restrict__ sums, int Nn, int D) {
  int id = blockIdx.x * 256 + threadIdx.x;
  if (id >= Nn * D) return;
  int i = id / D, f = id - i * D;
  atomicAdd(&sums[(size_t)batch[i] * D + f], h[(size_t)i * D + f]);
}

__global__ void k_cnt(const int* __restrict__ batch, float* __restrict__ cnt, int Nn) {
  int i = blockIdx.x * 256 + threadIdx.x;
  if (i < Nn) atomicAdd(&cnt[batch[i]], 1.0f);
}

__global__ void k_reparam(const float* __restrict__ summu, const float* __restrict__ sumlog,
                          const float* __restrict__ cnt, const float* __restrict__ noise,
                          float* __restrict__ z, float* __restrict__ omu,
                          float* __restrict__ olog, int Bn, int D) {
  int id = blockIdx.x * 256 + threadIdx.x;
  if (id >= Bn * D) return;
  int b = id / D;
  float c  = fmaxf(cnt[b], 1.0f);
  float mu = summu[id] / c;
  float lv = sumlog[id] / c;
  omu[id]  = mu;
  olog[id] = lv;
  z[id]    = mu + noise[id] * expf(0.5f * lv);
}

// ---------------- fused GIN MLP: out = act((relu((h+agg)W1+b1))W2 + b2) ----------------
// Block = 256 threads = 8 waves; each block handles a 64-row node tile; din,dh <= 64.
// Weights are staged TRANSPOSED ([N][K]) in LDS so B fragments are contiguous loads.
__global__ void __launch_bounds__(256)
k_gin(const float* __restrict__ hin, const float* __restrict__ agg,
      const float* __restrict__ w1, const float* __restrict__ b1,
      const float* __restrict__ w2, const float* __restrict__ b2,
      float* __restrict__ hout, int Nn, int din, int dh, int is_final) {
  __shared__ _Float16 sT  [64 * 64];   // (h+agg) tile, [64][din]
  __shared__ _Float16 sW1t[64 * 64];   // W1^T, [dh][din]
  __shared__ _Float16 sW2t[64 * 64];   // W2^T, [dh][dh]
  __shared__ _Float16 sH2 [64 * 64];   // hidden, [64][dh]
  __shared__ float sB1[64];
  __shared__ float sB2[64];

  const int tid  = threadIdx.x;
  const int wid  = tid >> 5;
  const int lane = tid & 31;
  const int m0   = blockIdx.x * 64;

  for (int i = tid; i < din * dh; i += 256) {      // coalesced read, transposed write
    int k = i / dh, n = i - k * dh;
    sW1t[n * din + k] = (_Float16)w1[i];
  }
  for (int i = tid; i < dh * dh; i += 256) {
    int k = i / dh, n = i - k * dh;
    sW2t[n * dh + k] = (_Float16)w2[i];
  }
  for (int i = tid; i < dh; i += 256) { sB1[i] = b1[i]; sB2[i] = b2[i]; }
  for (int i = tid; i < 64 * din; i += 256) {
    int r = i / din, c = i - r * din;
    int g = m0 + r;
    float v = 0.0f;
    if (g < Nn) v = hin[(size_t)g * din + c] + agg[(size_t)g * din + c];
    sT[i] = (_Float16)v;
  }
  __syncthreads();

  const int ct   = dh >> 4;     // 16-col tiles
  const int ntl  = 4 * ct;      // 16x16 tiles over 64 rows
  const int n16  = lane & 15;
  const int moff = (lane >> 4) * 8;

  // H2 = relu(T @ W1 + b1)
  for (int t = wid; t < ntl; t += 8) {
    int mt = t / ct, nt = t - mt * ct;
    v8f acc = {};
    for (int k0 = 0; k0 < din; k0 += 32) {
      v16h a = frag_a_lds(sT + mt * 16 * din + k0, din);
      v16h b = frag_b_t(sW1t + nt * 16 * din + k0, din);
      acc = __builtin_amdgcn_wmma_f32_16x16x32_f16(false, a, false, b, (short)0, acc, false, false);
    }
    float bb = sB1[nt * 16 + n16];
#pragma unroll
    for (int r = 0; r < 8; ++r) {
      float v = acc[r] + bb;
      v = v > 0.0f ? v : 0.0f;
      sH2[(mt * 16 + moff + r) * dh + nt * 16 + n16] = (_Float16)v;
    }
  }
  __syncthreads();

  // OUT = H2 @ W2 + b2 (+ relu unless final)
  for (int t = wid; t < ntl; t += 8) {
    int mt = t / ct, nt = t - mt * ct;
    v8f acc = {};
    for (int k0 = 0; k0 < dh; k0 += 32) {
      v16h a = frag_a_lds(sH2 + mt * 16 * dh + k0, dh);
      v16h b = frag_b_t(sW2t + nt * 16 * dh + k0, dh);
      acc = __builtin_amdgcn_wmma_f32_16x16x32_f16(false, a, false, b, (short)0, acc, false, false);
    }
    float bb = sB2[nt * 16 + n16];
#pragma unroll
    for (int r = 0; r < 8; ++r) {
      int g = m0 + mt * 16 + moff + r;
      if (g < Nn) {
        float v = acc[r] + bb;
        if (!is_final) v = v > 0.0f ? v : 0.0f;
        hout[(size_t)g * dh + nt * 16 + n16] = v;
      }
    }
  }
}

// ---------------- generic FC GEMM: Y = act(X @ W + b), M rows, K -> Nc ----------------
// grid = (ceil(Nc/128), M/16); block = 256 = 8 waves, one 16x16 tile per wave.
// Per K-step: stage X (16x32) and a 128x32 transposed W tile in LDS (coalesced reads).
__global__ void __launch_bounds__(256)
k_fc(const float* __restrict__ X, const float* __restrict__ W,
     const float* __restrict__ bias, float* __restrict__ Y,
     int M, int K, int Nc, int act) {
  __shared__ _Float16 sX [16 * 32];
  __shared__ _Float16 sWt[128 * 32];   // W^T tile: [n 0..127][k 0..31]
  const int tid  = threadIdx.x;
  const int wid  = tid >> 5;
  const int lane = tid & 31;
  const int m0   = blockIdx.y * 16;
  const int nb   = blockIdx.x * 128;

  v8f acc = {};
  for (int k0 = 0; k0 < K; k0 += 32) {
    __syncthreads();
    for (int i = tid; i < 16 * 32; i += 256) {
      int r = i >> 5, c = i & 31;
      int row = m0 + r;
      float v = (row < M) ? X[(size_t)row * K + k0 + c] : 0.0f;
      sX[i] = (_Float16)v;
    }
    for (int i = tid; i < 128 * 32; i += 256) {
      int k = i >> 7, n = i & 127;         // consecutive threads -> consecutive cols
      int gn = nb + n;
      float v = (gn < Nc) ? W[(size_t)(k0 + k) * Nc + gn] : 0.0f;
      sWt[n * 32 + k] = (_Float16)v;
    }
    __syncthreads();
    v16h a = frag_a_lds(sX, 32);
    v16h b = frag_b_t(sWt + wid * 16 * 32, 32);
    acc = __builtin_amdgcn_wmma_f32_16x16x32_f16(false, a, false, b, (short)0, acc, false, false);
  }

  int n = nb + wid * 16 + (lane & 15);
  if (n < Nc) {
    int moff = (lane >> 4) * 8;
    float bb = bias[n];
#pragma unroll
    for (int r = 0; r < 8; ++r) {
      int row = m0 + moff + r;
      if (row < M) {
        float v = acc[r] + bb;
        if (act == 1)      v = v > 0.0f ? v : 0.0f;
        else if (act == 2) v = 1.0f / (1.0f + expf(-v));
        Y[(size_t)row * Nc + n] = v;
      }
    }
  }
}

// ---------------- launch ----------------
extern "C" void kernel_launch(void* const* d_in, const int* in_sizes, int n_in,
                              void* d_out, int out_size, void* d_ws, size_t ws_size,
                              hipStream_t stream) {
  const int N = 50000, E = 800000, Bn = 256, DMAX = 64;

  const float* x     = (const float*)d_in[0];
  const int*   ei    = (const int*)d_in[1];
  const int*   src   = ei;          // ei[0]
  const int*   dst   = ei + E;      // ei[1]
  const int*   batch = (const int*)d_in[2];
  const float* noise = (const float*)d_in[3];

  float* ws     = (float*)d_ws;
  float* agg    = ws; ws += (size_t)N * DMAX;
  float* hA     = ws; ws += (size_t)N * DMAX;
  float* hB     = ws; ws += (size_t)N * DMAX;
  float* summu  = ws; ws += Bn * 64;
  float* sumlog = ws; ws += Bn * 64;
  float* cnt    = ws; ws += Bn;
  float* z      = ws; ws += Bn * 64;
  float* a1     = ws; ws += Bn * 128;
  float* a2     = ws; ws += Bn * 256;
  float* a3     = ws; ws += Bn * 512;

  float* out     = (float*)d_out;           // [256, 2964]
  float* out_mu  = out + (size_t)Bn * 2964; // [256, 64]
  float* out_log = out_mu + (size_t)Bn * 64;

  const int din_arr[4] = {64, 32, 64, 64};
  const int dh_arr[4]  = {32, 64, 64, 64};

  // zero pooling accumulators (summu|sumlog|cnt contiguous), then counts
  {
    int nz = Bn * 64 * 2 + Bn;
    k_zero<<<(nz + 255) / 256, 256, 0, stream>>>(summu, nz);
  }
  k_cnt<<<(N + 255) / 256, 256, 0, stream>>>(batch, cnt, N);

  // two encoder branches: mu (params at d_in[4..19]), log (d_in[20..35])
  for (int branch = 0; branch < 2; ++branch) {
    int base = 4 + branch * 16;
    const float* h = x;
    float* bufs[2] = {hA, hB};
    int pp = 0;
    for (int l = 0; l < 4; ++l) {
      const float* w1 = (const float*)d_in[base + 4 * l + 0];
      const float* b1 = (const float*)d_in[base + 4 * l + 1];
      const float* w2 = (const float*)d_in[base + 4 * l + 2];
      const float* b2 = (const float*)d_in[base + 4 * l + 3];
      int din = din_arr[l], dh = dh_arr[l];

      int na = N * din;
      k_zero<<<(na + 255) / 256, 256, 0, stream>>>(agg, na);
      int ne = E * din;
      k_agg<<<(ne + 255) / 256, 256, 0, stream>>>(h, src, dst, agg, E, din);

      float* ho = bufs[pp]; pp ^= 1;
      k_gin<<<(N + 63) / 64, 256, 0, stream>>>(h, agg, w1, b1, w2, b2, ho, N, din, dh,
                                               (l == 3) ? 1 : 0);
      h = ho;
    }
    float* sums = (branch == 0) ? summu : sumlog;
    int np = N * 64;
    k_pool<<<(np + 255) / 256, 256, 0, stream>>>(h, batch, sums, N, 64);
  }

  {
    int nr = Bn * 64;
    k_reparam<<<(nr + 255) / 256, 256, 0, stream>>>(summu, sumlog, cnt, noise,
                                                    z, out_mu, out_log, Bn, 64);
  }

  // decoder FC chain: 64 -> 128 -> 256 -> 512 -> 2964 (sigmoid)
  const float* fw[4]; const float* fb[4];
  for (int i = 0; i < 4; ++i) {
    fw[i] = (const float*)d_in[36 + 2 * i];
    fb[i] = (const float*)d_in[36 + 2 * i + 1];
  }
  const int fk[4] = {64, 128, 256, 512};
  const int fn[4] = {128, 256, 512, 2964};
  float* xs[5] = {z, a1, a2, a3, out};
  for (int i = 0; i < 4; ++i) {
    dim3 grid((fn[i] + 127) / 128, Bn / 16);
    k_fc<<<grid, 256, 0, stream>>>(xs[i], fw[i], fb[i], xs[i + 1],
                                   Bn, fk[i], fn[i], (i == 3) ? 2 : 1);
  }
}
